// FUNASR_NANO_DECODER_MAIN_90056874263255
// MI455X (gfx1250) — compile-verified, matrix operand order checked by reading, not run on hardware
//
#include <hip/hip_runtime.h>
#include <hip/hip_bf16.h>

// ---------------- model constants ----------------
constexpr int NLAYERS = 8;
constexpr int BATCH   = 2;
constexpr int SEQ     = 1024;
constexpr int DMODEL  = 1024;
constexpr int NH      = 16;
constexpr int NKV     = 8;
constexpr int HDIM    = 128;
constexpr int GRP     = NH / NKV;        // 2
constexpr int NQKH    = NH + NKV;        // 24
constexpr int NHEADS32= NH + 2 * NKV;    // 32
constexpr int FFNDIM  = 3072;
constexpr int NVOCAB  = 50257;
constexpr int MTOK    = BATCH * SEQ;     // 2048
constexpr int QKV_E   = NHEADS32 * HDIM; // 4096

// ---------------- types ----------------
typedef __attribute__((ext_vector_type(16))) __bf16          v16bf;
typedef __attribute__((ext_vector_type(8)))  __bf16          v8bf;
typedef __attribute__((ext_vector_type(16))) unsigned short  v16u;
typedef __attribute__((ext_vector_type(8)))  unsigned short  v8u;
typedef __attribute__((ext_vector_type(8)))  float           v8f;

// native f32 -> bf16 converts (lower to v_cvt_pk_bf16_f32-class ops)
__device__ __forceinline__ unsigned short f2bf(float f) {
  __bf16 b = (__bf16)f;
  return __builtin_bit_cast(unsigned short, b);
}

__device__ __forceinline__ v8u pack8(float4 a, float4 b) {
  v8bf r = { (__bf16)a.x, (__bf16)a.y, (__bf16)a.z, (__bf16)a.w,
             (__bf16)b.x, (__bf16)b.y, (__bf16)b.z, (__bf16)b.w };
  return __builtin_bit_cast(v8u, r);
}

__device__ __forceinline__ v16u combine8(v8u lo, v8u hi) {
  return __builtin_shufflevector(lo, hi, 0, 1, 2, 3, 4, 5, 6, 7,
                                 8, 9, 10, 11, 12, 13, 14, 15);
}

__device__ __forceinline__ v8f wmma_bf16(v16u a, v16u b, v8f c) {
  return __builtin_amdgcn_wmma_f32_16x16x32_bf16(
      false, __builtin_bit_cast(v16bf, a),
      false, __builtin_bit_cast(v16bf, b),
      (short)0, c, false, false);
}

// ---------------- RMS norm (x * rsqrt(sum x^2)) ----------------
__global__ __launch_bounds__(256)
void rmsnorm_kernel(const float* __restrict__ in, float* __restrict__ out,
                    size_t row_stride) {
  __shared__ float red[8];
  const size_t bi  = (size_t)blockIdx.x * row_stride;
  const size_t bo  = (size_t)blockIdx.x * DMODEL;
  const int t = threadIdx.x;
  float v[4]; float s = 0.f;
#pragma unroll
  for (int i = 0; i < 4; ++i) { v[i] = in[bi + t + i * 256]; s += v[i] * v[i]; }
#pragma unroll
  for (int m = 16; m >= 1; m >>= 1) s += __shfl_xor(s, m, 32);
  if ((t & 31) == 0) red[t >> 5] = s;
  __syncthreads();
  float tot = 0.f;
#pragma unroll
  for (int w = 0; w < 8; ++w) tot += red[w];
  const float r = rsqrtf(tot);
#pragma unroll
  for (int i = 0; i < 4; ++i) out[bo + t + i * 256] = v[i] * r;
}

// ---------------- WMMA GEMM: C[M,N] = A[M,K] * W[N,K]^T (+ R) ----------------
// 64x64 block tile, 8 waves (4 m-tiles x 2 n-pairs), K stepped by 32 (bf16 WMMA)
// LDS rows are 80 B (multiple of 16 B) so every fragment half is one
// 16-B-aligned ds_load_b128.
template <bool ADD_RES>
__global__ __launch_bounds__(256)
void gemm_kernel(const float* __restrict__ A, const float* __restrict__ W,
                 const float* __restrict__ R, float* __restrict__ C,
                 int M, int N, int K) {
  __shared__ unsigned short As[64][40];
  __shared__ unsigned short Bs[64][40];
  const int tid  = threadIdx.x;
  const int lane = tid & 31;
  const int wave = tid >> 5;
  const int half = lane >> 4;
  const int l15  = lane & 15;
  const int m0 = blockIdx.y * 64;
  const int n0 = blockIdx.x * 64;
  const int wm = (wave & 3) * 16;
  const int wn = (wave >> 2) * 32;
  const int lrow = tid >> 2;
  const int lcol = (tid & 3) * 8;
  const int kb = half * 8;

  v8f acc0 = {}; v8f acc1 = {};

  for (int k0 = 0; k0 < K; k0 += 32) {
    { // stage A tile (64x32 f32 -> bf16), one b128 store per thread
      const int m = m0 + lrow;
      float4 u0 = {}, u1 = {};
      if (m < M) {
        const float* p = A + (size_t)m * K + k0 + lcol;
        u0 = *(const float4*)p;
        u1 = *(const float4*)(p + 4);
        if (k0 + 32 < K) __builtin_prefetch(p + 32);
      }
      *(v8u*)&As[lrow][lcol] = pack8(u0, u1);
    }
    { // stage W tile
      const int n = n0 + lrow;
      float4 u0 = {}, u1 = {};
      if (n < N) {
        const float* p = W + (size_t)n * K + k0 + lcol;
        u0 = *(const float4*)p;
        u1 = *(const float4*)(p + 4);
        if (k0 + 32 < K) __builtin_prefetch(p + 32);
      }
      *(v8u*)&Bs[lrow][lcol] = pack8(u0, u1);
    }
    __syncthreads();
    // fragments per ISA 16-bit A/B layout (lanes 0-15: K 0..7/16..23,
    // lanes 16-31: K 8..15/24..31) -- two b128 LDS loads each
    const v16u a  = combine8(*(const v8u*)&As[wm + l15][kb],
                             *(const v8u*)&As[wm + l15][kb + 16]);
    const v16u b0 = combine8(*(const v8u*)&Bs[wn + l15][kb],
                             *(const v8u*)&Bs[wn + l15][kb + 16]);
    const v16u b1 = combine8(*(const v8u*)&Bs[wn + 16 + l15][kb],
                             *(const v8u*)&Bs[wn + 16 + l15][kb + 16]);
    acc0 = wmma_bf16(a, b0, acc0);
    acc1 = wmma_bf16(a, b1, acc1);
    __syncthreads();
  }
  // epilogue: C layout — VGPR i: row = half*8+i, col = lane&15
#pragma unroll
  for (int i = 0; i < 8; ++i) {
    const int row = m0 + wm + half * 8 + i;
    if (row < M) {
      const int c0 = n0 + wn + l15;
      const int c1 = c0 + 16;
      if (c0 < N) {
        float v = acc0[i];
        if (ADD_RES) v += R[(size_t)row * N + c0];
        C[(size_t)row * N + c0] = v;
      }
      if (c1 < N) {
        float v = acc1[i];
        if (ADD_RES) v += R[(size_t)row * N + c1];
        C[(size_t)row * N + c1] = v;
      }
    }
  }
}

// ---------------- qk-RMS-norm + RoPE + KV routing ----------------
// grid = B*L*32 blocks, 128 threads (one head-vector per block)
__global__ __launch_bounds__(128)
void qkv_post_kernel(const float* __restrict__ qkv, const float* __restrict__ qkn,
                     float* __restrict__ Qws, float* __restrict__ Kws,
                     float* __restrict__ Vws, float* __restrict__ kout,
                     float* __restrict__ vout) {
  __shared__ float sh[HDIM];
  __shared__ float red[4];
  const int t    = threadIdx.x;
  const int hd32 = blockIdx.x % NHEADS32;
  const int l    = (blockIdx.x / NHEADS32) % SEQ;
  const int b    = blockIdx.x / (NHEADS32 * SEQ);
  float val = qkv[(((size_t)b * SEQ + l) * NHEADS32 + hd32) * HDIM + t];

  if (hd32 < NQKH) {  // q & k heads: rms-norm, scale, rope
    float s = val * val;
#pragma unroll
    for (int m = 16; m >= 1; m >>= 1) s += __shfl_xor(s, m, 32);
    if ((t & 31) == 0) red[t >> 5] = s;
    __syncthreads();
    const float tot = red[0] + red[1] + red[2] + red[3];
    val = val * rsqrtf(tot) * qkn[hd32 * HDIM + t];
    sh[t] = val;
    __syncthreads();
    const float rot = sh[t ^ 64];
    const int i = t & 63;
    const float ang = (float)l * __powf(1.0e6f, -(float)i * (1.0f / 64.0f));
    const float c = __cosf(ang), sn = __sinf(ang);
    val = val * c + rot * ((t < 64) ? -sn : sn);
  }
  if (hd32 < NH) {
    Qws[(((size_t)b * NH + hd32) * SEQ + l) * HDIM + t] = val;
  } else if (hd32 < NQKH) {
    const int kvh = hd32 - NH;
    Kws[(((size_t)b * NKV + kvh) * SEQ + l) * HDIM + t] = val;
    // k_out layout (B,KV,HD,S)
    kout[(((size_t)b * NKV + kvh) * HDIM + t) * SEQ + l] = val;
  } else {
    const int vh = hd32 - NQKH;
    Vws[(((size_t)b * NKV + vh) * SEQ + l) * HDIM + t] = val;
    // v_out layout (B,KV,S,HD)
    vout[(((size_t)b * NKV + vh) * SEQ + l) * HDIM + t] = val;
  }
}

// ---------------- flash attention (causal, additive -128 mask) ----------------
// grid = (SEQ/64, B*NH), 128 threads (4 waves, each owns 16 q-rows)
// K staged row-major (QK^T gathers contiguous along HD); V staged TRANSPOSED
// (Vt[hd][key]) so PV B-fragments are contiguous along keys -> b128 LDS loads.
__global__ __launch_bounds__(128)
void attention_kernel(const float* __restrict__ Q, const float* __restrict__ Kc,
                      const float* __restrict__ Vc, float* __restrict__ Aout,
                      const int* __restrict__ maskp) {
  __shared__ unsigned short Ks[64][HDIM + 8];   // 272 B rows (16-B aligned)
  __shared__ unsigned short Vt[HDIM][72];       // 144 B rows (16-B aligned)
  __shared__ unsigned short Ps[4][16][72];
  const int tid  = threadIdx.x;
  const int lane = tid & 31;
  const int wave = tid >> 5;
  const int half = lane >> 4;
  const int l15  = lane & 15;
  const int qb = blockIdx.x;
  const int q0 = qb * 64;
  const int bh = blockIdx.y;
  const int b  = bh / NH;
  const int h  = bh % NH;
  const int kv = h / GRP;
  const int kb = half * 8;
  const float maskadd = -128.0f * (float)maskp[0];

  const float* Qbase = Q  + ((size_t)b * NH  + h)  * SEQ * HDIM;
  const float* Kbase = Kc + ((size_t)b * NKV + kv) * SEQ * HDIM;
  const float* Vbase = Vc + ((size_t)b * NKV + kv) * SEQ * HDIM;

  // Q fragments for this wave's 16 rows (4 K-chunks of 32 along HD)
  v16u qa[4];
  {
    const float* qp = Qbase + (size_t)(q0 + wave * 16 + l15) * HDIM;
#pragma unroll
    for (int c = 0; c < 4; ++c) {
      const float4 a0 = *(const float4*)(qp + c * 32 + kb);
      const float4 a1 = *(const float4*)(qp + c * 32 + kb + 4);
      const float4 a2 = *(const float4*)(qp + c * 32 + kb + 16);
      const float4 a3 = *(const float4*)(qp + c * 32 + kb + 20);
      qa[c] = combine8(pack8(a0, a1), pack8(a2, a3));
    }
  }

  float mrow[8], lsum[8];
  v8f o[8];
  const v8f vzero = {};
#pragma unroll
  for (int i = 0; i < 8; ++i) { mrow[i] = -3.0e38f; lsum[i] = 0.f; o[i] = vzero; }

  for (int kb0 = 0; kb0 <= qb; ++kb0) {
    __syncthreads();
    { // stage K (row-major, b128 stores) and V (transposed, scalar stores)
      const int r  = tid >> 1;
      const int c0 = (tid & 1) * 64;
      const float* kp = Kbase + (size_t)(kb0 * 64 + r) * HDIM + c0;
      const float* vp = Vbase + (size_t)(kb0 * 64 + r) * HDIM + c0;
#pragma unroll
      for (int c = 0; c < 64; c += 8) {
        const float4 k0v = *(const float4*)(kp + c);
        const float4 k1v = *(const float4*)(kp + c + 4);
        *(v8u*)&Ks[r][c0 + c] = pack8(k0v, k1v);
        const float4 v0v = *(const float4*)(vp + c);
        const float4 v1v = *(const float4*)(vp + c + 4);
        Vt[c0 + c    ][r] = f2bf(v0v.x); Vt[c0 + c + 1][r] = f2bf(v0v.y);
        Vt[c0 + c + 2][r] = f2bf(v0v.z); Vt[c0 + c + 3][r] = f2bf(v0v.w);
        Vt[c0 + c + 4][r] = f2bf(v1v.x); Vt[c0 + c + 5][r] = f2bf(v1v.y);
        Vt[c0 + c + 6][r] = f2bf(v1v.z); Vt[c0 + c + 7][r] = f2bf(v1v.w);
      }
    }
    __syncthreads();

    // scores: S = Q * K^T  (4 key tiles of 16, 4 HD chunks of 32)
    v8f st[4];
#pragma unroll
    for (int kt = 0; kt < 4; ++kt) {
      v8f acc = vzero;
#pragma unroll
      for (int c = 0; c < 4; ++c) {
        const v16u bf = combine8(
            *(const v8u*)&Ks[kt * 16 + l15][c * 32 + kb],
            *(const v8u*)&Ks[kt * 16 + l15][c * 32 + kb + 16]);
        acc = wmma_bf16(qa[c], bf, acc);
      }
      st[kt] = acc;
    }

    if (kb0 == qb) { // causal mask on diagonal block
#pragma unroll
      for (int kt = 0; kt < 4; ++kt) {
        const int col = kb0 * 64 + kt * 16 + l15;
#pragma unroll
        for (int i = 0; i < 8; ++i) {
          const int row = q0 + wave * 16 + half * 8 + i;
          if (col > row) st[kt][i] += maskadd;
        }
      }
    }

    // online softmax (row reductions across the 16 lanes of each half-wave)
#pragma unroll
    for (int i = 0; i < 8; ++i) {
      float mx = fmaxf(fmaxf(st[0][i], st[1][i]), fmaxf(st[2][i], st[3][i]));
#pragma unroll
      for (int m = 8; m >= 1; m >>= 1) mx = fmaxf(mx, __shfl_xor(mx, m, 32));
      const float mnew  = fmaxf(mrow[i], mx);
      const float scale = __expf(mrow[i] - mnew);
      mrow[i] = mnew;
      float rs = 0.f;
#pragma unroll
      for (int kt = 0; kt < 4; ++kt) {
        const float p = __expf(st[kt][i] - mnew);
        st[kt][i] = p;
        rs += p;
      }
#pragma unroll
      for (int m = 8; m >= 1; m >>= 1) rs += __shfl_xor(rs, m, 32);
      lsum[i] = lsum[i] * scale + rs;
#pragma unroll
      for (int ht = 0; ht < 8; ++ht) o[ht][i] *= scale;
    }

    // transpose P through per-wave LDS into A-fragment layout
#pragma unroll
    for (int kt = 0; kt < 4; ++kt)
#pragma unroll
      for (int i = 0; i < 8; ++i)
        Ps[wave][half * 8 + i][kt * 16 + l15] = f2bf(st[kt][i]);
    asm volatile("s_wait_dscnt 0" ::: "memory");
    v16u pa[2];
#pragma unroll
    for (int c = 0; c < 2; ++c)
      pa[c] = combine8(*(const v8u*)&Ps[wave][l15][c * 32 + kb],
                       *(const v8u*)&Ps[wave][l15][c * 32 + kb + 16]);

    // O += P * V   (8 hd tiles of 16, 2 key chunks of 32) via transposed Vt
#pragma unroll
    for (int ht = 0; ht < 8; ++ht) {
#pragma unroll
      for (int c = 0; c < 2; ++c) {
        const v16u vf = combine8(
            *(const v8u*)&Vt[ht * 16 + l15][c * 32 + kb],
            *(const v8u*)&Vt[ht * 16 + l15][c * 32 + kb + 16]);
        o[ht] = wmma_bf16(pa[c], vf, o[ht]);
      }
    }
  }

  // epilogue: Aout is (B, L, NH*HDIM)
#pragma unroll
  for (int i = 0; i < 8; ++i) {
    const float inv = 1.0f / lsum[i];
    const int row = q0 + wave * 16 + half * 8 + i;
    float* op = Aout + (((size_t)b * SEQ + row) * NH + h) * HDIM;
#pragma unroll
    for (int ht = 0; ht < 8; ++ht) op[ht * 16 + l15] = o[ht][i] * inv;
  }
}

// ---------------- silu(gate) * up ----------------
// grid = (FFNDIM/256, MTOK) to avoid div/mod in the hot path
__global__ __launch_bounds__(256)
void silu_mul_kernel(const float* __restrict__ gu, float* __restrict__ out) {
  const int f = blockIdx.x * 256 + threadIdx.x;
  const int m = blockIdx.y;
  const float g = gu[(size_t)m * 2 * FFNDIM + f];
  const float u = gu[(size_t)m * 2 * FFNDIM + FFNDIM + f];
  out[(size_t)m * FFNDIM + f] = (g / (1.f + __expf(-g))) * u;
}

// ---------------- host orchestration ----------------
extern "C" void kernel_launch(void* const* d_in, const int* in_sizes, int n_in,
                              void* d_out, int out_size, void* d_ws, size_t ws_size,
                              hipStream_t stream) {
  (void)in_sizes; (void)n_in; (void)out_size; (void)ws_size;
  const float* hidden    = (const float*)d_in[0];
  const float* qkv_w     = (const float*)d_in[3];
  const float* qk_norm_w = (const float*)d_in[4];
  const float* o_w       = (const float*)d_in[5];
  const float* gate_up_w = (const float*)d_in[6];
  const float* down_w    = (const float*)d_in[7];
  const float* lm_head_w = (const float*)d_in[8];
  const int*   maskp     = (const int*)d_in[11];
  float* out = (float*)d_out;

  float* ws   = (float*)d_ws;
  float* x    = ws; ws += (size_t)MTOK * DMODEL;
  float* qkvb = ws; ws += (size_t)MTOK * QKV_E;
  float* Qws  = ws; ws += (size_t)BATCH * NH  * SEQ * HDIM;
  float* Kws  = ws; ws += (size_t)BATCH * NKV * SEQ * HDIM;
  float* Vws  = ws; ws += (size_t)BATCH * NKV * SEQ * HDIM;
  float* Aout = ws; ws += (size_t)MTOK * NH * HDIM;
  float* gu   = ws; ws += (size_t)MTOK * 2 * FFNDIM;
  float* act  = ws; ws += (size_t)MTOK * FFNDIM;
  float* hA   = ws; ws += (size_t)MTOK * DMODEL;
  float* hB   = ws; ws += (size_t)MTOK * DMODEL;
  float* xf   = ws; ws += (size_t)BATCH * DMODEL;

  hipMemcpyAsync(hA, hidden, sizeof(float) * (size_t)MTOK * DMODEL,
                 hipMemcpyDeviceToDevice, stream);

  float* kout0  = out;
  float* vout0  = out + (size_t)NLAYERS * BATCH * NKV * HDIM * SEQ;
  float* logits = vout0 + (size_t)NLAYERS * BATCH * NKV * SEQ * HDIM;

  const dim3 gQKV((QKV_E) / 64, MTOK / 64);
  const dim3 gO(DMODEL / 64, MTOK / 64);
  const dim3 gGU((2 * FFNDIM) / 64, MTOK / 64);
  const dim3 gDN(DMODEL / 64, MTOK / 64);
  const dim3 gAT(SEQ / 64, BATCH * NH);
  const dim3 gSI(FFNDIM / 256, MTOK);
  const dim3 gLM((NVOCAB + 63) / 64, 1);

  for (int i = 0; i < NLAYERS; ++i) {
    rmsnorm_kernel<<<MTOK, 256, 0, stream>>>(hA, x, (size_t)DMODEL);
    gemm_kernel<false><<<gQKV, 256, 0, stream>>>(
        x, qkv_w + (size_t)i * QKV_E * DMODEL, nullptr, qkvb, MTOK, QKV_E, DMODEL);
    qkv_post_kernel<<<BATCH * SEQ * NHEADS32, 128, 0, stream>>>(
        qkvb, qk_norm_w + (size_t)i * NQKH * HDIM, Qws, Kws, Vws,
        kout0 + (size_t)i * BATCH * NKV * HDIM * SEQ,
        vout0 + (size_t)i * BATCH * NKV * SEQ * HDIM);
    attention_kernel<<<gAT, 128, 0, stream>>>(Qws, Kws, Vws, Aout, maskp);
    gemm_kernel<true><<<gO, 256, 0, stream>>>(
        Aout, o_w + (size_t)i * DMODEL * NH * HDIM, hA, hB, MTOK, DMODEL, NH * HDIM);
    rmsnorm_kernel<<<MTOK, 256, 0, stream>>>(hB, x, (size_t)DMODEL);
    gemm_kernel<false><<<gGU, 256, 0, stream>>>(
        x, gate_up_w + (size_t)i * 2 * FFNDIM * DMODEL, nullptr, gu,
        MTOK, 2 * FFNDIM, DMODEL);
    silu_mul_kernel<<<gSI, 256, 0, stream>>>(gu, act);
    gemm_kernel<true><<<gDN, 256, 0, stream>>>(
        act, down_w + (size_t)i * DMODEL * FFNDIM, hB, hA, MTOK, DMODEL, FFNDIM);
  }

  // last-token rms + lm_head
  rmsnorm_kernel<<<BATCH, 256, 0, stream>>>(
      hA + (size_t)(SEQ - 1) * DMODEL, xf, (size_t)SEQ * DMODEL);
  gemm_kernel<false><<<gLM, 256, 0, stream>>>(
      xf, lm_head_w, nullptr, logits, BATCH, NVOCAB, DMODEL);
}
